// Model_25881472926499
// MI455X (gfx1250) — matrix operand order repaired and checked by reading, not compiled, not run on hardware
//
#include <hip/hip_runtime.h>

// ---------------------------------------------------------------------------
// Model: 2-layer GRU encoder (L=168) -> 2-layer GRU decoder (Lf=24), H=32,
// B=32, D=N*C=1600, plus kNN gather and output projection.
// MI455X strategy:
//  * big input-gate GEMMs  -> v_wmma_f32_16x16x32_f16 (f16 in, f32 acc)
//  * latency-bound scans   -> single persistent workgroup, state in LDS,
//                             per-step matmul via v_wmma_f32_16x16x4_f32,
//                             gx_t double-buffered in LDS via async
//                             global->LDS loads (ASYNCcnt) to hide the
//                             per-step L2 round-trip behind the WMMAs.
// ---------------------------------------------------------------------------

typedef __attribute__((ext_vector_type(16))) _Float16 v16h;
typedef __attribute__((ext_vector_type(8)))  float    v8f;
typedef __attribute__((ext_vector_type(2)))  float    v2f;
typedef __attribute__((ext_vector_type(4)))  int      v4i;

#define B_   32
#define C_   8
#define N_   200
#define L_   168
#define LF_  24
#define D_   1600
#define H_   32
#define G3_  96      // 3*H
#define GRID_ 4096   // 64*64 era grid points (last col dropped)

#if defined(__has_builtin)
#if __has_builtin(__builtin_amdgcn_wmma_f32_16x16x4_f32)
#define HAVE_WMMA_F32X4 1
#endif
#if __has_builtin(__builtin_amdgcn_global_load_async_to_lds_b128) && \
    __has_builtin(__builtin_amdgcn_s_wait_asynccnt)
#define HAVE_ASYNC_LDS 1
#endif
#endif

typedef __attribute__((address_space(1))) v4i gv4i;   // global 16B chunk
typedef __attribute__((address_space(3))) v4i lv4i;   // LDS 16B chunk

__device__ __forceinline__ float sigm_f(float x) {
  return 1.0f / (1.0f + __expf(-x));
}
__device__ __forceinline__ float tanh_f(float x) {
  // tanh(x) = 1 - 2/(exp(2x)+1); saturates correctly at +/-inf.
  return 1.0f - 2.0f / (__expf(2.0f * x) + 1.0f);
}

// ---------------------------------------------------------------------------
// kNN: for each station, argmin over 64x64 era grid (first-min tie break).
// ---------------------------------------------------------------------------
__global__ void knn_kernel(const float* __restrict__ csta,
                           const float* __restrict__ cera,
                           int* __restrict__ idx) {
  int n = blockIdx.x * blockDim.x + threadIdx.x;
  if (n >= N_) return;
  float sx = csta[n * 2 + 0], sy = csta[n * 2 + 1];
  float best = 3.4e38f;
  int bi = 0;
  for (int gh = 0; gh < 64; ++gh) {
    for (int gw = 0; gw < 64; ++gw) {
      float dx = sx - cera[(gh * 65 + gw) * 2 + 0];
      float dy = sy - cera[(gh * 65 + gw) * 2 + 1];
      float d = dx * dx + dy * dy;
      int g = gh * 64 + gw;
      if (d < best) { best = d; bi = g; }
    }
  }
  idx[n] = bi;
}

// ---------------------------------------------------------------------------
// f32 -> f16 weight convert (runs every launch; tiny).
// ---------------------------------------------------------------------------
__global__ void cvt_f16_kernel(const float* __restrict__ src,
                               _Float16* __restrict__ dst, int n) {
  int i = blockIdx.x * blockDim.x + threadIdx.x;
  if (i < n) dst[i] = (_Float16)src[i];
}

// ---------------------------------------------------------------------------
// WMMA f16 fragment layouts (per CDNA5 ISA, wave32):
// A (16x32, M x K): lane holds row M = lane&15.
//   lane<16: K = {0..7, 16..23};  lane>=16: K = {8..15, 24..31}  (pairs/VGPR)
// B (32x16, K x N): lane holds column N = lane&15.
//   ISA: same K split as A; this code loads A and B with lane<16: K=0..15, lane>=16: K=16..31
// C (16x16 f32): lane<16: N=lane, M=v; lane>=16: N=lane-16, M=v+8.
// ---------------------------------------------------------------------------

// Epilogue: store C tile + bias to gx[(row)*96 + col]
__device__ __forceinline__ void store_c_tile(float* __restrict__ gx,
                                             const v8f& acc, float bias,
                                             int mtile, int ncol, int lane) {
  int mbase = mtile * 16 + ((lane & 16) ? 8 : 0);
#pragma unroll
  for (int v = 0; v < 8; ++v)
    gx[(size_t)(mbase + v) * G3_ + ncol] = acc[v] + bias;
}

// ---------------------------------------------------------------------------
// gx_enc0 = X @ Wih0^T + bih0 ; X gathered from obs_his on the fly.
// Rows are time-major: r = l*32 + b  (so the scan reads contiguous steps).
// One wave computes one 16-row M-tile across all 6 N-tiles (96 cols).
// grid: 42 blocks * 8 waves = 336 M-tiles = 5376 rows.
// ---------------------------------------------------------------------------
__global__ void __launch_bounds__(256)
gemm_gx_obs(const float* __restrict__ obs,      // (B,C,N,L)
            const _Float16* __restrict__ Wh,    // (96,1600) f16
            const float* __restrict__ bias,     // (96)
            float* __restrict__ gx) {           // (L*B, 96)
  const int lane = threadIdx.x & 31;
  const int wave = threadIdx.x >> 5;
  const int mtile = blockIdx.x * 8 + wave;
  const int r = mtile * 16 + (lane & 15);       // global row for A gather
  const int b = r & 31, l = r >> 5;
  const int grp = (lane & 16) ? 8 : 0;
  const int khalf = (lane & 16) ? 16 : 0;
  const int ncl = lane & 15;

  v8f acc[6] = {};
  for (int kb = 0; kb < D_; kb += 32) {
    // gather A fragment (16 scalars, L2-resident after first touch)
    v16h a;
#pragma unroll
    for (int j = 0; j < 16; ++j) {
      int k = kb + khalf + j;
      int n = k >> 3, c = k & 7;
      float x = obs[((size_t)(b * C_ + c) * N_ + n) * L_ + l];
      a[j] = (_Float16)x;
    }
#pragma unroll
    for (int nt = 0; nt < 6; ++nt) {
      int ncol = nt * 16 + ncl;
      v16h bfrag = *(const v16h*)(&Wh[(size_t)ncol * D_ + kb + khalf]);
      acc[nt] = __builtin_amdgcn_wmma_f32_16x16x32_f16(
          false, a, false, bfrag, (short)0, acc[nt], false, false);
    }
  }
#pragma unroll
  for (int nt = 0; nt < 6; ++nt) {
    int ncol = nt * 16 + ncl;
    store_c_tile(gx, acc[nt], bias[ncol], mtile, ncol, lane);
  }
}

// ---------------------------------------------------------------------------
// gx (layer1) = ys @ Wih1^T + bih1 ; ys is (rows, 32) f32, K=32 single step.
// ---------------------------------------------------------------------------
__global__ void __launch_bounds__(256)
gemm_gx_h(const float* __restrict__ ys,        // (rows, 32)
          const _Float16* __restrict__ Wh,     // (96,32) f16
          const float* __restrict__ bias,
          float* __restrict__ gx) {
  const int lane = threadIdx.x & 31;
  const int wave = threadIdx.x >> 5;
  const int mtile = blockIdx.x * 8 + wave;
  const int r = mtile * 16 + (lane & 15);
  const int grp = (lane & 16) ? 8 : 0;
  const int khalf = (lane & 16) ? 16 : 0;
  const int ncl = lane & 15;

  v16h a;
#pragma unroll
  for (int j = 0; j < 16; ++j) {
    int k = khalf + j;
    a[j] = (_Float16)ys[(size_t)r * H_ + k];
  }
#pragma unroll
  for (int nt = 0; nt < 6; ++nt) {
    int ncol = nt * 16 + ncl;
    v16h bfrag = *(const v16h*)(&Wh[(size_t)ncol * H_ + khalf]);
    v8f acc = {};
    acc = __builtin_amdgcn_wmma_f32_16x16x32_f16(
        false, a, false, bfrag, (short)0, acc, false, false);
    store_c_tile(gx, acc, bias[ncol], mtile, ncol, lane);
  }
}

// ---------------------------------------------------------------------------
// gx_dec0 = DecIn @ Wih0d^T + bih0d ; DecIn gathered from pan_fut via idx.
// rows = Lf*B = 768 -> 48 tiles -> 6 blocks.
// ---------------------------------------------------------------------------
__global__ void __launch_bounds__(256)
gemm_gx_pan(const float* __restrict__ pan,     // (B,C,4096,Lf)
            const int* __restrict__ idx,       // (N)
            const _Float16* __restrict__ Wh,   // (96,1600) f16
            const float* __restrict__ bias,
            float* __restrict__ gx) {          // (Lf*B, 96)
  __shared__ int sidx[N_];
  for (int i = threadIdx.x; i < N_; i += blockDim.x) sidx[i] = idx[i];
  __syncthreads();

  const int lane = threadIdx.x & 31;
  const int wave = threadIdx.x >> 5;
  const int mtile = blockIdx.x * 8 + wave;
  const int r = mtile * 16 + (lane & 15);
  const int b = r & 31, t = r >> 5;
  const int grp = (lane & 16) ? 8 : 0;
  const int khalf = (lane & 16) ? 16 : 0;
  const int ncl = lane & 15;

  v8f acc[6] = {};
  for (int kb = 0; kb < D_; kb += 32) {
    v16h a;
#pragma unroll
    for (int j = 0; j < 16; ++j) {
      int k = kb + khalf + j;
      int n = k >> 3, c = k & 7;
      float x = pan[((size_t)(b * C_ + c) * GRID_ + sidx[n]) * LF_ + t];
      a[j] = (_Float16)x;
    }
#pragma unroll
    for (int nt = 0; nt < 6; ++nt) {
      int ncol = nt * 16 + ncl;
      v16h bfrag = *(const v16h*)(&Wh[(size_t)ncol * D_ + kb + khalf]);
      acc[nt] = __builtin_amdgcn_wmma_f32_16x16x32_f16(
          false, a, false, bfrag, (short)0, acc[nt], false, false);
    }
  }
#pragma unroll
  for (int nt = 0; nt < 6; ++nt) {
    int ncol = nt * 16 + ncl;
    store_c_tile(gx, acc[nt], bias[ncol], mtile, ncol, lane);
  }
}

// ---------------------------------------------------------------------------
// GRU scan: one persistent workgroup (12 waves = 384 threads).
// Per step: gh(32x96) = h(32x32) @ Whh^T + bhh via 12 WMMA f32 16x16x4
// tiles (full fp32 recurrence), then fused gate elementwise + h update.
// gx_t is double-buffered in LDS with async global->LDS loads so the fetch
// for step t+1 overlaps step t's gate update and step t+1's WMMAs.
// ---------------------------------------------------------------------------
__global__ void __launch_bounds__(384)
gru_scan(const float* __restrict__ gx,   // (T, 32, 96)
         const float* __restrict__ Whh,  // (96, 32)
         const float* __restrict__ bhh,  // (96)
         const float* __restrict__ h0,   // (32,32) or nullptr -> zeros
         float* __restrict__ ys,         // (T,32,32) or nullptr
         float* __restrict__ hT,         // (32,32)
         int T) {
  __shared__ float sW[G3_ * H_];   // Whh
  __shared__ float sb[G3_];        // bhh
  __shared__ float sh[B_ * H_];    // hidden state
  __shared__ float sgh[B_ * G3_];  // gh staging
#if HAVE_ASYNC_LDS
  __shared__ float sgx[2][B_ * G3_];  // double-buffered gate inputs
#endif

  const int tid = threadIdx.x;
  for (int i = tid; i < G3_ * H_; i += 384) sW[i] = Whh[i];
  if (tid < G3_) sb[tid] = bhh[tid];
  for (int i = tid; i < B_ * H_; i += 384) sh[i] = h0 ? h0[i] : 0.0f;

#if HAVE_ASYNC_LDS
  // Each of the 384 threads async-copies 2x16B: 768 * 16B = 12KB = one step.
  {
    const float* src0 = gx;  // step 0
#pragma unroll
    for (int s2 = 0; s2 < 2; ++s2) {
      int eo = (s2 * 384 + tid) * 4;  // float offset, 16B aligned
      __builtin_amdgcn_global_load_async_to_lds_b128(
          (gv4i*)(src0 + eo), (lv4i*)(&sgx[0][eo]), 0, 0);
    }
  }
#endif
  __syncthreads();

  const int lane = tid & 31;
  const int wave = tid >> 5;        // 0..11
  const int mt = wave / 6;          // 0..1  (batch tile)
  const int nt = wave % 6;          // 0..5  (gate tile)
  const int ncl = lane & 15;
  const int ncol = nt * 16 + ncl;
#if HAVE_WMMA_F32X4
  const int mrow = mt * 16 + ncl;          // A row for this lane
  const int klo = (lane & 16) ? 2 : 0;     // A/B K sub-offset per lane half
  const int mbase = mt * 16 + ((lane & 16) ? 8 : 0);
#endif

  for (int t = 0; t < T; ++t) {
    // ---- gh = h @ Whh^T + bhh -------------------------------------------
#if HAVE_WMMA_F32X4
    float bb = sb[ncol];
    v8f c;
#pragma unroll
    for (int v = 0; v < 8; ++v) c[v] = bb;
#pragma unroll
    for (int kb = 0; kb < H_; kb += 4) {
      v2f a, b;
      a.x = sh[mrow * H_ + kb + klo];
      a.y = sh[mrow * H_ + kb + klo + 1];
      b.x = sW[ncol * H_ + kb + klo];
      b.y = sW[ncol * H_ + kb + klo + 1];
      c = __builtin_amdgcn_wmma_f32_16x16x4_f32(
          false, a, false, b, (short)0, c, false, false);
    }
#pragma unroll
    for (int v = 0; v < 8; ++v) sgh[(mbase + v) * G3_ + ncol] = c[v];
#else
    for (int e = tid; e < B_ * G3_; e += 384) {
      int j = e / G3_, g = e % G3_;
      float acc = sb[g];
#pragma unroll 8
      for (int k = 0; k < H_; ++k) acc += sh[j * H_ + k] * sW[g * H_ + k];
      sgh[e] = acc;
    }
    (void)mt; (void)nt; (void)ncol;
#endif

#if HAVE_ASYNC_LDS
    // Prefetch gx for step t+1 into the other buffer, then wait only for
    // the current step's data (allow the 2 just-issued loads in flight).
    if (t + 1 < T) {
      const float* srcn = gx + (size_t)(t + 1) * (B_ * G3_);
      int bufn = (t + 1) & 1;
#pragma unroll
      for (int s2 = 0; s2 < 2; ++s2) {
        int eo = (s2 * 384 + tid) * 4;
        __builtin_amdgcn_global_load_async_to_lds_b128(
            (gv4i*)(srcn + eo), (lv4i*)(&sgx[bufn][eo]), 0, 0);
      }
      __builtin_amdgcn_s_wait_asynccnt(2);
    } else {
      __builtin_amdgcn_s_wait_asynccnt(0);
    }
#endif
    __syncthreads();

    // ---- gates + hidden update ------------------------------------------
#if HAVE_ASYNC_LDS
    const float* gxt = &sgx[t & 1][0];
#else
    const float* gxt = gx + (size_t)t * B_ * G3_;
#endif
    for (int e = tid; e < B_ * H_; e += 384) {
      int j = e >> 5, i = e & 31;
      float r = sigm_f(gxt[j * G3_ + i]          + sgh[j * G3_ + i]);
      float z = sigm_f(gxt[j * G3_ + H_ + i]     + sgh[j * G3_ + H_ + i]);
      float nn = tanh_f(gxt[j * G3_ + 2 * H_ + i] + r * sgh[j * G3_ + 2 * H_ + i]);
      float hn = (1.0f - z) * nn + z * sh[e];
      sh[e] = hn;
      if (ys) ys[(size_t)t * (B_ * H_) + e] = hn;
    }
    __syncthreads();
  }
  for (int e = tid; e < B_ * H_; e += 384) hT[e] = sh[e];
}

// ---------------------------------------------------------------------------
// y[b,0,n,t] = dot(ys_dec1[t,b,:], outW[n,:]) + outb[n]
// ---------------------------------------------------------------------------
__global__ void __launch_bounds__(256)
out_proj(const float* __restrict__ ys,    // (Lf,32,32)
         const float* __restrict__ outW,  // (200,32)
         const float* __restrict__ outb,  // (200)
         float* __restrict__ y) {         // (B,1,N,Lf) flat
  int e = blockIdx.x * blockDim.x + threadIdx.x;
  if (e >= B_ * N_ * LF_) return;
  int t = e % LF_;
  int n = (e / LF_) % N_;
  int b = e / (LF_ * N_);
  const float* h = ys + (size_t)(t * B_ + b) * H_;
  const float* w = outW + (size_t)n * H_;
  float acc = outb[n];
#pragma unroll 8
  for (int k = 0; k < H_; ++k) acc += h[k] * w[k];
  y[e] = acc;
}

// ---------------------------------------------------------------------------
extern "C" void kernel_launch(void* const* d_in, const int* in_sizes, int n_in,
                              void* d_out, int out_size, void* d_ws,
                              size_t ws_size, hipStream_t stream) {
  const float* obs   = (const float*)d_in[0];
  // d_in[1] era_his: unused by the reference.
  const float* pan   = (const float*)d_in[2];
  const float* csta  = (const float*)d_in[3];
  const float* cera  = (const float*)d_in[4];
  // d_in[5] cpan: unused.
  const float* eWih0 = (const float*)d_in[6];
  const float* eWhh0 = (const float*)d_in[7];
  const float* ebih0 = (const float*)d_in[8];
  const float* ebhh0 = (const float*)d_in[9];
  const float* eWih1 = (const float*)d_in[10];
  const float* eWhh1 = (const float*)d_in[11];
  const float* ebih1 = (const float*)d_in[12];
  const float* ebhh1 = (const float*)d_in[13];
  const float* dWih0 = (const float*)d_in[14];
  const float* dWhh0 = (const float*)d_in[15];
  const float* dbih0 = (const float*)d_in[16];
  const float* dbhh0 = (const float*)d_in[17];
  const float* dWih1 = (const float*)d_in[18];
  const float* dWhh1 = (const float*)d_in[19];
  const float* dbih1 = (const float*)d_in[20];
  const float* dbhh1 = (const float*)d_in[21];
  const float* outW  = (const float*)d_in[22];
  const float* outb  = (const float*)d_in[23];

  char* ws = (char*)d_ws;
  size_t off = 0;
  auto alloc = [&](size_t bytes) -> void* {
    off = (off + 255) & ~(size_t)255;
    void* p = ws + off;
    off += bytes;
    return p;
  };

  int*       idx      = (int*)alloc(N_ * sizeof(int));
  _Float16*  eWih0h   = (_Float16*)alloc((size_t)G3_ * D_ * 2);
  _Float16*  dWih0h   = (_Float16*)alloc((size_t)G3_ * D_ * 2);
  _Float16*  eWih1h   = (_Float16*)alloc((size_t)G3_ * H_ * 2);
  _Float16*  dWih1h   = (_Float16*)alloc((size_t)G3_ * H_ * 2);
  float*     gx_enc0  = (float*)alloc((size_t)L_ * B_ * G3_ * 4);
  float*     ys_enc0  = (float*)alloc((size_t)L_ * B_ * H_ * 4);
  float*     gx_enc1  = (float*)alloc((size_t)L_ * B_ * G3_ * 4);
  float*     gx_dec0  = (float*)alloc((size_t)LF_ * B_ * G3_ * 4);
  float*     ys_dec0  = (float*)alloc((size_t)LF_ * B_ * H_ * 4);
  float*     gx_dec1  = (float*)alloc((size_t)LF_ * B_ * G3_ * 4);
  float*     ys_dec1  = (float*)alloc((size_t)LF_ * B_ * H_ * 4);
  float*     h_enc0   = (float*)alloc((size_t)B_ * H_ * 4);
  float*     h_enc1   = (float*)alloc((size_t)B_ * H_ * 4);
  float*     h_scr0   = (float*)alloc((size_t)B_ * H_ * 4);
  float*     h_scr1   = (float*)alloc((size_t)B_ * H_ * 4);
  (void)ws_size; (void)in_sizes; (void)n_in; (void)out_size;

  // Independent prep
  knn_kernel<<<1, 256, 0, stream>>>(csta, cera, idx);
  cvt_f16_kernel<<<600, 256, 0, stream>>>(eWih0, eWih0h, G3_ * D_);
  cvt_f16_kernel<<<600, 256, 0, stream>>>(dWih0, dWih0h, G3_ * D_);
  cvt_f16_kernel<<<12, 256, 0, stream>>>(eWih1, eWih1h, G3_ * H_);
  cvt_f16_kernel<<<12, 256, 0, stream>>>(dWih1, dWih1h, G3_ * H_);

  // Encoder
  gemm_gx_obs<<<42, 256, 0, stream>>>(obs, eWih0h, ebih0, gx_enc0);
  gru_scan<<<1, 384, 0, stream>>>(gx_enc0, eWhh0, ebhh0, nullptr,
                                  ys_enc0, h_enc0, L_);
  gemm_gx_h<<<42, 256, 0, stream>>>(ys_enc0, eWih1h, ebih1, gx_enc1);
  gru_scan<<<1, 384, 0, stream>>>(gx_enc1, eWhh1, ebhh1, nullptr,
                                  nullptr, h_enc1, L_);

  // Decoder
  gemm_gx_pan<<<6, 256, 0, stream>>>(pan, idx, dWih0h, dbih0, gx_dec0);
  gru_scan<<<1, 384, 0, stream>>>(gx_dec0, dWhh0, dbhh0, h_enc0,
                                  ys_dec0, h_scr0, LF_);
  gemm_gx_h<<<6, 256, 0, stream>>>(ys_dec0, dWih1h, dbih1, gx_dec1);
  gru_scan<<<1, 384, 0, stream>>>(gx_dec1, dWhh1, dbhh1, h_enc1,
                                  ys_dec1, h_scr1, LF_);

  // Output projection
  out_proj<<<600, 256, 0, stream>>>(ys_dec1, outW, outb, (float*)d_out);
}
